// GRUCritic_63367947485746
// MI455X (gfx1250) — matrix operand order repaired and checked by reading, not compiled, not run on hardware
//
#include <hip/hip_runtime.h>

// Fused GRU critic for MI455X (gfx1250, wave32).
//
// Roofline: state_seq = 536 MB must be read exactly once (~23 us at 23.3 TB/s).
// Weights (~100 KB) are converted to f16 once and live in LDS / VGPRs; all
// per-step activations stay in registers with tiny LDS transpose bounces.
// One wave owns a 16-row batch tile and runs the full T=512 scan; all three
// GEMMs per step use v_wmma_f32_16x16x32_f16 (f16 in, f32 accumulate):
// 64 WMMA / step / wave. 128 independent waves, one per block, so each wave
// lands on its own WGP (full LDS bandwidth, 1 wave/SIMD WMMA issue regime).

typedef __attribute__((ext_vector_type(16))) _Float16 v16h;
typedef __attribute__((ext_vector_type(8)))  _Float16 v8h;
typedef __attribute__((ext_vector_type(8)))  float    v8f;
typedef __attribute__((ext_vector_type(4)))  float    v4f;

namespace {
constexpr int kBatch = 2048;
constexpr int kTsteps = 512;
constexpr int kD = 128;   // STATE_DIM
constexpr int kH = 64;    // HIDDEN == GRU_H
constexpr int kG = 192;   // 3 * GRU_H
constexpr int kRows = 16;              // batch rows per wave (WMMA M)
constexpr int kTiles = 64;             // 16 (W1) + 24 (W_ih) + 24 (W_hh)
}

__device__ __forceinline__ v8f zero8() {
  v8f z = {0.f, 0.f, 0.f, 0.f, 0.f, 0.f, 0.f, 0.f};
  return z;
}
__device__ __forceinline__ float fast_sigmoid(float x) {
  return 1.0f / (1.0f + __expf(-x));
}
__device__ __forceinline__ float fast_tanh(float x) {
  float e = __expf(2.0f * x);
  return (e - 1.0f) / (e + 1.0f);
}

__global__ __launch_bounds__(32, 1)
void gru_critic_fused(const float* __restrict__ state,
                      const float* __restrict__ W1,
                      const float* __restrict__ b1,
                      const float* __restrict__ W_ih,
                      const float* __restrict__ W_hh,
                      const float* __restrict__ b_ih,
                      const float* __restrict__ b_hh,
                      const float* __restrict__ W_out,
                      const float* __restrict__ b_out,
                      float* __restrict__ out) {
  // Weight tiles pre-swizzled into per-lane WMMA B-operand order:
  // wt[tile][lane] holds 16 f16 = B[k][n], n = 16*nt+(lane&15),
  // k = 32*kt + 8*(lane>>4) + (e<8 ? e : 8+e). One tile fetch = 2x ds_load_b128
  // (the compiler is free to promote hot tiles into high VGPRs).
  __shared__ v16h     wt[kTiles][32];                 // 64 KB
  __shared__ _Float16 xstage[kRows][kH];              // 2 KB (D->A transpose bounce)
  __shared__ _Float16 hstage[2][kRows][kH];           // 4 KB (double buffered)
  __shared__ float bias1[kH];
  __shared__ float biasIH[kG];
  __shared__ float biasHH[kG];

  const int tid  = threadIdx.x;
  const int lane = tid & 31;
  const int n    = lane & 15;   // WMMA column within 16-wide tile / A row id
  const int h2   = lane >> 4;   // which half-wave

  // ---- one-time: convert f32 weights -> f16 WMMA B layout in LDS ----
  for (int slot = tid; slot < kTiles * 32; slot += 32) {
    const int tile = slot >> 5;
    const int sl   = slot & 31;
    const int sn   = sl & 15;
    const int sh   = sl >> 4;
    const float* W;
    int nt, kt, ldk;
    if (tile < 16)      { kt = tile >> 2;  nt = tile & 3;              W = W1;   ldk = kD; }
    else if (tile < 40) { int u = tile - 16; kt = u / 12; nt = u % 12; W = W_ih; ldk = kH; }
    else                { int u = tile - 40; kt = u / 12; nt = u % 12; W = W_hh; ldk = kH; }
    const int nG = 16 * nt + sn;
    v16h v;
#pragma unroll
    for (int e = 0; e < 16; ++e) {
      const int kG2 = 32 * kt + 8 * sh + (e < 8 ? e : 8 + e);    // B layout: K = 8*(lane>>4)+(e<8 ? e : 8+e)
      v[e] = (_Float16)W[nG * ldk + kG2];       // B[k][n] = W[n][k] (D = A * W^T)
    }
    wt[tile][sl] = v;
  }
  for (int i = tid; i < kH; i += 32) bias1[i] = b1[i];
  for (int i = tid; i < kG; i += 32) { biasIH[i] = b_ih[i]; biasHH[i] = b_hh[i]; }
  {
    _Float16* hz = &hstage[0][0][0];
    for (int i = tid; i < kRows * kH; i += 32) hz[i] = (_Float16)0.0f;
  }
  __syncthreads();   // single wave: cheap; orders LDS staging before use

  const int b0 = blockIdx.x * kRows;
  const float* srow = state + (size_t)(b0 + n) * kTsteps * kD;   // row m = lane&15

  v8f hreg[4];   // h in D layout: hreg[jt][r] = h[r + 8*h2][16*jt + n]
#pragma unroll
  for (int jt = 0; jt < 4; ++jt) hreg[jt] = zero8();

  for (int t = 0; t < kTsteps; ++t) {
    const float* sp = srow + (size_t)t * kD;
    if (t + 1 < kTsteps)
      __builtin_prefetch(srow + (size_t)(t + 1) * kD + 64 * h2, 0, 3);
    const int p = t & 1;        // read h buffer
    const int q = p ^ 1;        // write h buffer

    // ---- x = relu(s @ W1^T + b1) : [16 x 64], K = 128 ----
    v8f xacc[4];
#pragma unroll
    for (int nt = 0; nt < 4; ++nt) xacc[nt] = zero8();
#pragma unroll
    for (int kt = 0; kt < 4; ++kt) {
      // 16-bit A layout: e<8 -> K=32kt+8h2+e ; e>=8 -> K=32kt+8h2+16+(e-8)
      const float* pk = sp + 32 * kt + 8 * h2;
      const v4f f0 = *reinterpret_cast<const v4f*>(pk);
      const v4f f1 = *reinterpret_cast<const v4f*>(pk + 4);
      const v4f f2 = *reinterpret_cast<const v4f*>(pk + 16);
      const v4f f3 = *reinterpret_cast<const v4f*>(pk + 20);
      v16h a;
#pragma unroll
      for (int e = 0; e < 4; ++e) {
        a[e]      = (_Float16)f0[e];
        a[e + 4]  = (_Float16)f1[e];
        a[e + 8]  = (_Float16)f2[e];
        a[e + 12] = (_Float16)f3[e];
      }
#pragma unroll
      for (int nt = 0; nt < 4; ++nt)
        xacc[nt] = __builtin_amdgcn_wmma_f32_16x16x32_f16(
            false, a, false, wt[kt * 4 + nt][lane], (short)0, xacc[nt], false, false);
    }
    // bias + relu, D layout -> row-major f16 stage for next GEMM's A operand
#pragma unroll
    for (int nt = 0; nt < 4; ++nt) {
      const float bb = bias1[16 * nt + n];
#pragma unroll
      for (int r = 0; r < 8; ++r) {
        const float v = xacc[nt][r] + bb;
        xstage[r + 8 * h2][16 * nt + n] = (_Float16)fmaxf(v, 0.0f);
      }
    }

    // ---- per 16-wide hidden chunk: gx, gh, gate math, h update ----
#pragma unroll
    for (int jt = 0; jt < 4; ++jt) {
      v8f gxr = zero8(), gxz = zero8(), gxn = zero8();
      v8f ghr = zero8(), ghz = zero8(), ghn = zero8();
#pragma unroll
      for (int kt = 0; kt < 2; ++kt) {
        v16h ax, ah;
        {
          const _Float16* xb = &xstage[n][32 * kt + 8 * h2];
          const v8h lo = *reinterpret_cast<const v8h*>(xb);
          const v8h hi = *reinterpret_cast<const v8h*>(xb + 16);
#pragma unroll
          for (int e = 0; e < 8; ++e) { ax[e] = lo[e]; ax[e + 8] = hi[e]; }
        }
        {
          const _Float16* hb = &hstage[p][n][32 * kt + 8 * h2];
          const v8h lo = *reinterpret_cast<const v8h*>(hb);
          const v8h hi = *reinterpret_cast<const v8h*>(hb + 16);
#pragma unroll
          for (int e = 0; e < 8; ++e) { ah[e] = lo[e]; ah[e + 8] = hi[e]; }
        }
        gxr = __builtin_amdgcn_wmma_f32_16x16x32_f16(false, ax, false, wt[16 + kt * 12 + jt][lane],     (short)0, gxr, false, false);
        gxz = __builtin_amdgcn_wmma_f32_16x16x32_f16(false, ax, false, wt[16 + kt * 12 + jt + 4][lane], (short)0, gxz, false, false);
        gxn = __builtin_amdgcn_wmma_f32_16x16x32_f16(false, ax, false, wt[16 + kt * 12 + jt + 8][lane], (short)0, gxn, false, false);
        ghr = __builtin_amdgcn_wmma_f32_16x16x32_f16(false, ah, false, wt[40 + kt * 12 + jt][lane],     (short)0, ghr, false, false);
        ghz = __builtin_amdgcn_wmma_f32_16x16x32_f16(false, ah, false, wt[40 + kt * 12 + jt + 4][lane], (short)0, ghz, false, false);
        ghn = __builtin_amdgcn_wmma_f32_16x16x32_f16(false, ah, false, wt[40 + kt * 12 + jt + 8][lane], (short)0, ghn, false, false);
      }
      const int col = 16 * jt + n;
      const float bir = biasIH[col], biz = biasIH[col + 64], bin_ = biasIH[col + 128];
      const float bhr = biasHH[col], bhz = biasHH[col + 64], bhn = biasHH[col + 128];
#pragma unroll
      for (int r = 0; r < 8; ++r) {
        const float rr = fast_sigmoid(gxr[r] + bir + ghr[r] + bhr);
        const float zz = fast_sigmoid(gxz[r] + biz + ghz[r] + bhz);
        const float nn = fast_tanh(gxn[r] + bin_ + rr * (ghn[r] + bhn));
        const float hn = (1.0f - zz) * nn + zz * hreg[jt][r];
        hreg[jt][r] = hn;
        hstage[q][r + 8 * h2][col] = (_Float16)hn;
      }
    }
  }

  // ---- value = h_final @ W_out^T + b_out ----
  float wo[4];
#pragma unroll
  for (int jt = 0; jt < 4; ++jt) wo[jt] = W_out[16 * jt + n];
  const float bo = b_out[0];
#pragma unroll
  for (int r = 0; r < 8; ++r) {
    float acc = 0.0f;
#pragma unroll
    for (int jt = 0; jt < 4; ++jt) acc += hreg[jt][r] * wo[jt];
#pragma unroll
    for (int m = 1; m < 16; m <<= 1) acc += __shfl_xor(acc, m, 32);
    if (n == 0) out[b0 + r + 8 * h2] = acc + bo;
  }
}

extern "C" void kernel_launch(void* const* d_in, const int* in_sizes, int n_in,
                              void* d_out, int out_size, void* d_ws, size_t ws_size,
                              hipStream_t stream) {
  (void)in_sizes; (void)n_in; (void)out_size; (void)d_ws; (void)ws_size;
  const float* state = (const float*)d_in[0];
  const float* W1    = (const float*)d_in[1];
  const float* b1    = (const float*)d_in[2];
  const float* W_ih  = (const float*)d_in[3];
  const float* W_hh  = (const float*)d_in[4];
  const float* b_ih  = (const float*)d_in[5];
  const float* b_hh  = (const float*)d_in[6];
  const float* W_out = (const float*)d_in[7];
  const float* b_out = (const float*)d_in[8];
  float* out = (float*)d_out;

  dim3 grid(kBatch / kRows);   // 128 blocks -> one wave per WGP
  dim3 block(32);              // single wave32
  gru_critic_fused<<<grid, block, 0, stream>>>(state, W1, b1, W_ih, W_hh,
                                               b_ih, b_hh, W_out, b_out, out);
}
